// RXNG2Sequencer_37520834298522
// MI455X (gfx1250) — matrix operand order repaired and checked by reading, not compiled, not run on hardware
//
#include <hip/hip_runtime.h>
#include <hip/hip_bf16.h>
#include <math.h>

// Problem constants (match reference)
#define L_   4
#define D_   512
#define H_   8
#define DFF_ 2048
#define T_   256
#define B_   8
#define NB_  33          // BUCKETS+1
#define DH_  64
#define M_   (B_*T_)     // 2048 rows of activations

typedef _Float16 half_t;
typedef _Float16 v16h __attribute__((ext_vector_type(16)));
typedef _Float16 v8h  __attribute__((ext_vector_type(8)));
typedef float    v8f  __attribute__((ext_vector_type(8)));
typedef int      v4i  __attribute__((vector_size(16)));   // matches builtin param type

// Async global->LDS staging (CDNA5 GLOBAL_LOAD_ASYNC_TO_LDS_B128), with a
// synchronous fallback so the TU compiles on toolchains without the builtin.
#if defined(__gfx1250__) && __has_builtin(__builtin_amdgcn_global_load_async_to_lds_b128)
#define USE_ASYNC_LDS 1
typedef __attribute__((address_space(1))) v4i* gptr_t;
typedef __attribute__((address_space(3))) v4i* lptr_t;
#endif

// ---------------------------------------------------------------------------
// x[b,t,d] = src[t,b,d]*sqrt(D) + sinusoidal PE
// ---------------------------------------------------------------------------
__global__ void embed_kernel(const float* __restrict__ src, float* __restrict__ x) {
    int idx = blockIdx.x * blockDim.x + threadIdx.x;
    if (idx >= M_ * D_) return;
    int d   = idx % D_;
    int rid = idx / D_;
    int t   = rid % T_;
    int b   = rid / T_;
    float ang = (float)t * expf(-(float)(2 * (d / 2)) * (logf(10000.0f) / (float)D_));
    float pe  = (d & 1) ? cosf(ang) : sinf(ang);
    x[idx] = src[((size_t)t * B_ + b) * D_ + d] * 22.62741699796952f + pe; // sqrt(512)
}

// ---------------------------------------------------------------------------
// LayerNorm over D=512, one block (256 thr) per row, f16 output for WMMA
// ---------------------------------------------------------------------------
__global__ void ln_f16_kernel(const float* __restrict__ in, const float* __restrict__ g,
                              const float* __restrict__ be, half_t* __restrict__ outh) {
    int row = blockIdx.x;
    const float* r = in + (size_t)row * D_;
    __shared__ float red[256];
    int t = threadIdx.x;
    float a0 = r[t], a1 = r[t + 256];
    red[t] = a0 + a1; __syncthreads();
    for (int o = 128; o > 0; o >>= 1) { if (t < o) red[t] += red[t + o]; __syncthreads(); }
    float mean = red[0] * (1.0f / D_); __syncthreads();
    float d0 = a0 - mean, d1 = a1 - mean;
    red[t] = d0 * d0 + d1 * d1; __syncthreads();
    for (int o = 128; o > 0; o >>= 1) { if (t < o) red[t] += red[t + o]; __syncthreads(); }
    float inv = rsqrtf(red[0] * (1.0f / D_) + 1e-6f);
    outh[(size_t)row * D_ + t]       = (half_t)(d0 * inv * g[t] + be[t]);
    outh[(size_t)row * D_ + t + 256] = (half_t)(d1 * inv * g[t + 256] + be[t + 256]);
}

// Final LN: f32 output with (B,T,D) -> (T,B,D) transpose
__global__ void final_ln_kernel(const float* __restrict__ in, const float* __restrict__ g,
                                const float* __restrict__ be, float* __restrict__ out) {
    int row = blockIdx.x;              // row = b*T + t
    int b = row / T_, tt = row % T_;
    const float* r = in + (size_t)row * D_;
    __shared__ float red[256];
    int t = threadIdx.x;
    float a0 = r[t], a1 = r[t + 256];
    red[t] = a0 + a1; __syncthreads();
    for (int o = 128; o > 0; o >>= 1) { if (t < o) red[t] += red[t + o]; __syncthreads(); }
    float mean = red[0] * (1.0f / D_); __syncthreads();
    float d0 = a0 - mean, d1 = a1 - mean;
    red[t] = d0 * d0 + d1 * d1; __syncthreads();
    for (int o = 128; o > 0; o >>= 1) { if (t < o) red[t] += red[t + o]; __syncthreads(); }
    float inv = rsqrtf(red[0] * (1.0f / D_) + 1e-6f);
    float* o_ = out + ((size_t)tt * B_ + b) * D_;
    o_[t]       = d0 * inv * g[t] + be[t];
    o_[t + 256] = d1 * inv * g[t + 256] + be[t + 256];
}

__global__ void f32_to_f16_kernel(const float* __restrict__ in, half_t* __restrict__ out, int n) {
    int i = blockIdx.x * 256 + threadIdx.x;
    if (i < n) out[i] = (half_t)in[i];
}

// ---------------------------------------------------------------------------
// Split heads; build qu (f16), qv (f32), k (f16) as (B,H,T,DH); v transposed
// to (B,H,DH,T) f16 so attn@V fits the A@Bw^T GEMM pattern.
// ---------------------------------------------------------------------------
__global__ void prep_attn_kernel(const float* __restrict__ qraw, const float* __restrict__ kraw,
                                 const float* __restrict__ vraw, const float* __restrict__ u,
                                 const float* __restrict__ v, half_t* __restrict__ qu_h,
                                 float* __restrict__ qv, half_t* __restrict__ k_h,
                                 half_t* __restrict__ vT_h) {
    int idx = blockIdx.x * 256 + threadIdx.x;
    if (idx >= M_ * D_) return;
    int d   = idx % D_;
    int rid = idx / D_;
    int t   = rid % T_;
    int b   = rid / T_;
    int h = d / DH_, dh = d % DH_;
    float qs = qraw[idx] * 0.125f;                       // 1/sqrt(DH)
    size_t o = (((size_t)(b * H_ + h) * T_) + t) * DH_ + dh;
    qu_h[o] = (half_t)(qs + u[d]);
    qv[o]   = qs + v[d];
    k_h[o]  = (half_t)kraw[idx];
    vT_h[(((size_t)(b * H_ + h) * DH_) + dh) * T_ + t] = (half_t)vraw[idx];
}

// p[b,h,q,bucket] = (q/sqrt(dh)+v) . rel_table[bucket, head slice]
__global__ void rel_p_kernel(const float* __restrict__ qv, const float* __restrict__ rel,
                             float* __restrict__ p) {
    int idx = blockIdx.x * 256 + threadIdx.x;
    if (idx >= B_ * H_ * T_ * NB_) return;
    int bucket = idx % NB_;
    int r      = idx / NB_;            // (b*H+h)*T + q
    int h      = (r / T_) % H_;
    const float* q  = qv  + (size_t)r * DH_;
    const float* rr = rel + (size_t)bucket * D_ + h * DH_;
    float s = 0.f;
    #pragma unroll
    for (int i = 0; i < DH_; i++) s += q[i] * rr[i];
    p[idx] = s;
}

// scores += gathered rel term, mask, softmax -> attn (f16)
__global__ void softmax_kernel(const float* __restrict__ scores, const float* __restrict__ p,
                               const int* __restrict__ distances, const int* __restrict__ lengths,
                               half_t* __restrict__ attn_h) {
    int row = blockIdx.x;              // (b*H+h)*T + q
    int q   = row % T_;
    int b   = (row / T_) / H_;
    int k   = threadIdx.x;
    int dist = distances[((size_t)b * T_ + q) * T_ + k];
    float s = scores[(size_t)row * T_ + k] + p[(size_t)row * NB_ + dist];
    if (k >= lengths[b]) s = -1e30f;
    __shared__ float red[256];
    red[k] = s; __syncthreads();
    for (int o = 128; o > 0; o >>= 1) { if (k < o) red[k] = fmaxf(red[k], red[k + o]); __syncthreads(); }
    float mx = red[0]; __syncthreads();
    float e = expf(s - mx);
    red[k] = e; __syncthreads();
    for (int o = 128; o > 0; o >>= 1) { if (k < o) red[k] += red[k + o]; __syncthreads(); }
    attn_h[(size_t)row * T_ + k] = (half_t)(e / red[0]);
}

// (B,H,T,DH) f32 -> (B,T,D) f16 merged heads
__global__ void merge_ctx_kernel(const float* __restrict__ ctx, half_t* __restrict__ ctxm_h) {
    int idx = blockIdx.x * 256 + threadIdx.x;
    if (idx >= M_ * D_) return;
    int d   = idx % D_;
    int rid = idx / D_;
    int t   = rid % T_;
    int b   = rid / T_;
    int h = d / DH_, dh = d % DH_;
    ctxm_h[idx] = (half_t)ctx[(((size_t)(b * H_ + h) * T_) + t) * DH_ + dh];
}

// ---------------------------------------------------------------------------
// Batched WMMA GEMM:  C[z] = A[z](MxK) @ Bw[z](NxK)^T  (+bias +residual, relu)
// 256 threads = 8 waves; block tile 256(M) x 64(N); K-strip 32 via LDS.
// Each wave owns a 32x64 sub-tile (acc[2][4]) so every B fragment feeds two
// WMMAs. Tiles staged with GLOBAL_LOAD_ASYNC_TO_LDS_B128 when available.
// Fragments match CDNA5 ISA 7.12.2 layouts for V_WMMA_F32_16X16X32_F16.
// ---------------------------------------------------------------------------
__launch_bounds__(256)
__global__ void wmma_gemm_kernel(const half_t* __restrict__ A, long long lda, long long strideA,
                                 const half_t* __restrict__ Bw, long long ldb, long long strideB,
                                 float* __restrict__ C, long long ldc, long long strideC,
                                 half_t* __restrict__ Ch, long long ldch, long long strideCh,
                                 const float* __restrict__ bias,
                                 const float* __restrict__ Res, long long ldr, long long strideR,
                                 int K, int relu) {
    __shared__ __align__(16) half_t As[256 * 32];   // 16 KB
    __shared__ __align__(16) half_t Bs[64 * 32];    //  4 KB
    const int tid  = threadIdx.x;
    const int wave = tid >> 5;
    const int lane = tid & 31;
    const int m16  = lane & 15;        // row/col within 16 for fragments
    const int hsel = lane >> 4;        // lane-half selector
    const int z    = blockIdx.z;
    const long long m0 = (long long)blockIdx.y * 256;
    const long long n0 = (long long)blockIdx.x * 64;
    const half_t* Ag = A  + (long long)z * strideA + m0 * lda;
    const half_t* Bg = Bw + (long long)z * strideB + n0 * ldb;

    v8f acc[2][4] = {};

    for (int k0 = 0; k0 < K; k0 += 32) {
#ifdef USE_ASYNC_LDS
        // Async DMA: global -> LDS without touching VGPRs (ASYNCcnt-tracked)
        #pragma unroll
        for (int i = 0; i < 4; i++) {
            int lin = (i * 256 + tid) * 8;
            int r = lin >> 5, c = lin & 31;
            __builtin_amdgcn_global_load_async_to_lds_b128(
                (gptr_t)&Ag[(long long)r * lda + k0 + c],
                (lptr_t)&As[lin], 0, 0);
        }
        {
            int lin = tid * 8;
            int r = lin >> 5, c = lin & 31;
            __builtin_amdgcn_global_load_async_to_lds_b128(
                (gptr_t)&Bg[(long long)r * ldb + k0 + c],
                (lptr_t)&Bs[lin], 0, 0);
        }
#if __has_builtin(__builtin_amdgcn_s_wait_asynccnt)
        __builtin_amdgcn_s_wait_asynccnt(0);
#else
        asm volatile("s_wait_asynccnt 0" ::: "memory");
#endif
#else
        // Fallback: cooperative 16B register staging
        #pragma unroll
        for (int i = 0; i < 4; i++) {
            int lin = (i * 256 + tid) * 8;
            int r = lin >> 5, c = lin & 31;
            *(v8h*)&As[lin] = *(const v8h*)&Ag[(long long)r * lda + k0 + c];
        }
        {
            int lin = tid * 8;
            int r = lin >> 5, c = lin & 31;
            *(v8h*)&Bs[lin] = *(const v8h*)&Bg[(long long)r * ldb + k0 + c];
        }
#endif
        __syncthreads();

        // Two A fragments (16x32 each): lanes 0-15 K=0..7/16..23, lanes 16-31 K=8..15/24..31
        v16h a[2];
        #pragma unroll
        for (int s = 0; s < 2; s++) {
            const half_t* arow = &As[(wave * 32 + s * 16 + m16) * 32];
            v8h a0 = *(const v8h*)&arow[8 * hsel];
            v8h a1 = *(const v8h*)&arow[16 + 8 * hsel];
            #pragma unroll
            for (int e = 0; e < 8; e++) { a[s][e] = a0[e]; a[s][e + 8] = a1[e]; }
        }

        #pragma unroll
        for (int g = 0; g < 4; g++) {
            // B fragment (32x16): lane col = m16, K = 8*hsel + e / 16 + 8*hsel + e
            const half_t* brow = &Bs[(g * 16 + m16) * 32 + 8 * hsel];
            v8h b0 = *(const v8h*)&brow[0];
            v8h b1 = *(const v8h*)&brow[16];
            v16h bb;
            #pragma unroll
            for (int e = 0; e < 8; e++) { bb[e] = b0[e]; bb[e + 8] = b1[e]; }
            #pragma unroll
            for (int s = 0; s < 2; s++) {
                acc[s][g] = __builtin_amdgcn_wmma_f32_16x16x32_f16(
                    false, a[s], false, bb, (short)0, acc[s][g], false, false);
            }
        }
        __syncthreads();
    }

    // Epilogue: C/D layout -> m = r + 8*hsel (+s*16 +wave*32), n = m16 (+16*g)
    #pragma unroll
    for (int s = 0; s < 2; s++) {
        #pragma unroll
        for (int g = 0; g < 4; g++) {
            long long n = n0 + g * 16 + m16;
            float bv = bias ? bias[n] : 0.f;
            #pragma unroll
            for (int r = 0; r < 8; r++) {
                long long m = m0 + wave * 32 + s * 16 + r + 8 * hsel;
                float vv = acc[s][g][r] + bv;
                if (Res)  vv += Res[(long long)z * strideR + m * ldr + n];
                if (relu) vv = fmaxf(vv, 0.f);
                if (C)    C[(long long)z * strideC + m * ldc + n] = vv;
                if (Ch)   Ch[(long long)z * strideCh + m * ldch + n] = (half_t)vv;
            }
        }
    }
}

// ---------------------------------------------------------------------------
extern "C" void kernel_launch(void* const* d_in, const int* in_sizes, int n_in,
                              void* d_out, int out_size, void* d_ws, size_t ws_size,
                              hipStream_t stream) {
    (void)in_sizes; (void)n_in; (void)out_size; (void)ws_size;
    const float* src       = (const float*)d_in[0];
    const int*   lengths   = (const int*)  d_in[1];
    const int*   distances = (const int*)  d_in[2];
    const float* u         = (const float*)d_in[3];
    const float* v         = (const float*)d_in[4];
    const float* ln1_g     = (const float*)d_in[5];
    const float* ln1_b     = (const float*)d_in[6];
    const float* Wq        = (const float*)d_in[7];
    const float* Wk        = (const float*)d_in[8];
    const float* Wv        = (const float*)d_in[9];
    const float* Wo        = (const float*)d_in[10];
    const float* bq        = (const float*)d_in[11];
    const float* bk        = (const float*)d_in[12];
    const float* bv        = (const float*)d_in[13];
    const float* bo        = (const float*)d_in[14];
    const float* rel       = (const float*)d_in[15];
    const float* ffg       = (const float*)d_in[16];
    const float* ffb       = (const float*)d_in[17];
    const float* W1        = (const float*)d_in[18];
    const float* b1        = (const float*)d_in[19];
    const float* W2        = (const float*)d_in[20];
    const float* b2        = (const float*)d_in[21];
    const float* flg       = (const float*)d_in[22];
    const float* flb       = (const float*)d_in[23];

    char* ws = (char*)d_ws;
    size_t off = 0;
    auto alloc = [&](size_t bytes) -> char* {
        char* p = ws + off; off += (bytes + 255) & ~(size_t)255; return p;
    };
    float*  x        = (float*) alloc((size_t)M_ * D_ * 4);
    float*  outb     = (float*) alloc((size_t)M_ * D_ * 4);
    float*  qraw     = (float*) alloc((size_t)M_ * D_ * 4);
    float*  kraw     = (float*) alloc((size_t)M_ * D_ * 4);
    float*  vraw     = (float*) alloc((size_t)M_ * D_ * 4);
    float*  qv       = (float*) alloc((size_t)M_ * D_ * 4);
    float*  ctxb     = (float*) alloc((size_t)M_ * D_ * 4);
    float*  scores   = (float*) alloc((size_t)B_ * H_ * T_ * T_ * 4);
    float*  pbuf     = (float*) alloc((size_t)B_ * H_ * T_ * NB_ * 4);
    half_t* hln_h    = (half_t*)alloc((size_t)M_ * D_ * 2);
    half_t* h2_h     = (half_t*)alloc((size_t)M_ * D_ * 2);
    half_t* ctxm_h   = (half_t*)alloc((size_t)M_ * D_ * 2);
    half_t* qu_h     = (half_t*)alloc((size_t)M_ * D_ * 2);
    half_t* k_h      = (half_t*)alloc((size_t)M_ * D_ * 2);
    half_t* vT_h     = (half_t*)alloc((size_t)M_ * D_ * 2);
    half_t* attn_h   = (half_t*)alloc((size_t)B_ * H_ * T_ * T_ * 2);
    half_t* ffn1_h   = (half_t*)alloc((size_t)M_ * DFF_ * 2);
    half_t* Wqh      = (half_t*)alloc((size_t)D_ * D_ * 2);
    half_t* Wkh      = (half_t*)alloc((size_t)D_ * D_ * 2);
    half_t* Wvh      = (half_t*)alloc((size_t)D_ * D_ * 2);
    half_t* Woh      = (half_t*)alloc((size_t)D_ * D_ * 2);
    half_t* W1h      = (half_t*)alloc((size_t)DFF_ * D_ * 2);
    half_t* W2h      = (half_t*)alloc((size_t)D_ * DFF_ * 2);

    const int ew = (M_ * D_ + 255) / 256;
    embed_kernel<<<ew, 256, 0, stream>>>(src, x);

    for (int l = 0; l < L_; l++) {
        ln_f16_kernel<<<M_, 256, 0, stream>>>(x, ln1_g + l * D_, ln1_b + l * D_, hln_h);

        const int nDD = D_ * D_, nFD = DFF_ * D_;
        f32_to_f16_kernel<<<(nDD + 255) / 256, 256, 0, stream>>>(Wq + (size_t)l * nDD, Wqh, nDD);
        f32_to_f16_kernel<<<(nDD + 255) / 256, 256, 0, stream>>>(Wk + (size_t)l * nDD, Wkh, nDD);
        f32_to_f16_kernel<<<(nDD + 255) / 256, 256, 0, stream>>>(Wv + (size_t)l * nDD, Wvh, nDD);
        f32_to_f16_kernel<<<(nDD + 255) / 256, 256, 0, stream>>>(Wo + (size_t)l * nDD, Woh, nDD);
        f32_to_f16_kernel<<<(nFD + 255) / 256, 256, 0, stream>>>(W1 + (size_t)l * nFD, W1h, nFD);
        f32_to_f16_kernel<<<(nFD + 255) / 256, 256, 0, stream>>>(W2 + (size_t)l * nFD, W2h, nFD);

        // Q/K/V projections: (2048 x 512) @ (512 x 512)^T
        dim3 gDD(D_ / 64, M_ / 256, 1);
        wmma_gemm_kernel<<<gDD, 256, 0, stream>>>(hln_h, D_, 0, Wqh, D_, 0,
            qraw, D_, 0, (half_t*)nullptr, 0, 0, bq + l * D_, (float*)nullptr, 0, 0, D_, 0);
        wmma_gemm_kernel<<<gDD, 256, 0, stream>>>(hln_h, D_, 0, Wkh, D_, 0,
            kraw, D_, 0, (half_t*)nullptr, 0, 0, bk + l * D_, (float*)nullptr, 0, 0, D_, 0);
        wmma_gemm_kernel<<<gDD, 256, 0, stream>>>(hln_h, D_, 0, Wvh, D_, 0,
            vraw, D_, 0, (half_t*)nullptr, 0, 0, bv + l * D_, (float*)nullptr, 0, 0, D_, 0);

        prep_attn_kernel<<<ew, 256, 0, stream>>>(qraw, kraw, vraw, u, v, qu_h, qv, k_h, vT_h);
        rel_p_kernel<<<(B_ * H_ * T_ * NB_ + 255) / 256, 256, 0, stream>>>(
            qv, rel + (size_t)l * NB_ * D_, pbuf);

        // Content scores, batched over (b,h): (256x64) @ (256x64)^T
        dim3 gS(T_ / 64, T_ / 256, B_ * H_);
        wmma_gemm_kernel<<<gS, 256, 0, stream>>>(qu_h, DH_, (long long)T_ * DH_,
            k_h, DH_, (long long)T_ * DH_, scores, T_, (long long)T_ * T_,
            (half_t*)nullptr, 0, 0, (float*)nullptr, (float*)nullptr, 0, 0, DH_, 0);

        softmax_kernel<<<B_ * H_ * T_, 256, 0, stream>>>(scores, pbuf, distances, lengths, attn_h);

        // ctx = attn @ V, batched: (256x256) @ (64x256)^T  (Bw = V^T)
        dim3 gC(DH_ / 64, T_ / 256, B_ * H_);
        wmma_gemm_kernel<<<gC, 256, 0, stream>>>(attn_h, T_, (long long)T_ * T_,
            vT_h, T_, (long long)DH_ * T_, ctxb, DH_, (long long)T_ * DH_,
            (half_t*)nullptr, 0, 0, (float*)nullptr, (float*)nullptr, 0, 0, T_, 0);

        merge_ctx_kernel<<<ew, 256, 0, stream>>>(ctxb, ctxm_h);

        // Output projection + residual x
        wmma_gemm_kernel<<<gDD, 256, 0, stream>>>(ctxm_h, D_, 0, Woh, D_, 0,
            outb, D_, 0, (half_t*)nullptr, 0, 0, bo + l * D_, x, D_, 0, D_, 0);

        // FFN
        ln_f16_kernel<<<M_, 256, 0, stream>>>(outb, ffg + l * D_, ffb + l * D_, h2_h);
        dim3 gF1(DFF_ / 64, M_ / 256, 1);
        wmma_gemm_kernel<<<gF1, 256, 0, stream>>>(h2_h, D_, 0, W1h, D_, 0,
            (float*)nullptr, 0, 0, ffn1_h, DFF_, 0, b1 + l * DFF_, (float*)nullptr, 0, 0, D_, 1);
        wmma_gemm_kernel<<<gDD, 256, 0, stream>>>(ffn1_h, DFF_, 0, W2h, DFF_, 0,
            x, D_, 0, (half_t*)nullptr, 0, 0, b2 + l * D_, outb, D_, 0, DFF_, 0);
    }

    final_ln_kernel<<<M_, 256, 0, stream>>>(x, flg, flb, (float*)d_out);
}
